// MRN4GRUCell_84713934946666
// MI455X (gfx1250) — matrix operand order repaired and checked by reading, not compiled, not run on hardware
//
#include <hip/hip_runtime.h>

// MI455X / gfx1250 (wave32, WMMA). Roofline: ~305MB HBM traffic -> ~13us floor at
// 23.3 TB/s; compute reduced to ~29 GFLOPs via selected-type bucketing, run on
// v_wmma_f32_16x16x32_bf16 (f32 accum). Phase-2 B-slices are staged into LDS with
// double buffering, using GLOBAL_LOAD_ASYNC_TO_LDS_B128 + s_wait_asynccnt when the
// toolchain exposes the builtins. Requires ws_size >= ~1.2 MB (bf16 weights).

typedef __bf16 bf16_t;
typedef __attribute__((ext_vector_type(16))) __bf16 v16bf;
typedef __attribute__((ext_vector_type(8)))  float  v8f;

#define BATCH 65536
#define ISZ   64
#define HSZ   64
#define TNUM  16
#define G3H   192      // 3*H gate width
#define KCORE 1024     // T*H
#define ROWS  128      // batch rows per block
#define NWAVES 8
#define BPITCH 40      // LDS B-tile pitch in bf16 (80B: 16B-aligned, conflict-free)

#if defined(__gfx1250__) && \
    __has_builtin(__builtin_amdgcn_global_load_async_to_lds_b128) && \
    __has_builtin(__builtin_amdgcn_s_wait_asynccnt)
#define ASYNC_LDS 1
// Builtin signature (from clang diagnostic): param0 = addrspace(1) <4 x i32>*,
// param1 = LDS pointer, then imm offset + imm cpol.
typedef int v4i_ __attribute__((vector_size(16)));
typedef __attribute__((address_space(1))) v4i_ as1_v4i;
typedef __attribute__((address_space(3))) v4i_ as3_v4i;
#else
#define ASYNC_LDS 0
#endif

__device__ __forceinline__ v8f wmma_bf16(v16bf a, v16bf b, v8f c) {
  // D = A(16x32 bf16) * B(32x16 bf16) + C(16x16 f32)
  return __builtin_amdgcn_wmma_f32_16x16x32_bf16(false, a, false, b, (short)0, c,
                                                 false, false);
}

__device__ __forceinline__ v8f vzero8() {
  v8f z;
#pragma unroll
  for (int i = 0; i < 8; ++i) z[i] = 0.f;
  return z;
}

// A-fragment (16x32 bf16): lane<16 -> K {0..7,16..23}; lane>=16 -> K {8..15,24..31}
// base points at this lane's row at the K-chunk start; off = (lane<16)?0:8.
__device__ __forceinline__ v16bf afrag_f32(const float* base, int off) {
  v16bf a;
#pragma unroll
  for (int i = 0; i < 8; ++i) {
    a[i]     = (bf16_t)base[off + i];
    a[i + 8] = (bf16_t)base[16 + off + i];
  }
  return a;
}

__device__ __forceinline__ v16bf afrag_bf16(const bf16_t* base, int off) {
  v16bf a;
#pragma unroll
  for (int i = 0; i < 8; ++i) {
    a[i]     = base[off + i];
    a[i + 8] = base[16 + off + i];
  }
  return a;
}

// B-fragment (32x16 bf16): lane holds column N=lane&15, K {off..off+7, 16+off..16+off+7}.
__device__ __forceinline__ v16bf bfrag_bf16(const bf16_t* p) {
  union { uint4 q[2]; v16bf v; } u;
  u.q[0] = *(const uint4*)(p);
  u.q[1] = *(const uint4*)(p + 16);
  return u.v;
}

// 16B global -> LDS copy; async (ASYNCcnt) when available.
__device__ __forceinline__ void cp16(const bf16_t* gsrc, bf16_t* ldst) {
#if ASYNC_LDS
  __builtin_amdgcn_global_load_async_to_lds_b128((as1_v4i*)gsrc, (as3_v4i*)ldst,
                                                 0, 0);
#else
  *(uint4*)ldst = *(const uint4*)gsrc;
#endif
}

__device__ __forceinline__ void wait_cp_barrier() {
#if ASYNC_LDS
  __builtin_amdgcn_s_wait_asynccnt(0);
#endif
  __syncthreads();
}

__device__ __forceinline__ float sigm(float x) { return 1.f / (1.f + __expf(-x)); }

__global__ void cvt_bf16_kernel(const float* __restrict__ s, bf16_t* __restrict__ d, int n) {
  int i = blockIdx.x * 256 + threadIdx.x;
  if (i < n) d[i] = (bf16_t)s[i];
}

__global__ __launch_bounds__(256)
void mrn_gru_kernel(const float* __restrict__ x,
                    const int*   __restrict__ typ,
                    const float* __restrict__ cvec,
                    const float* __restrict__ regs,
                    const float* __restrict__ bih,
                    const float* __restrict__ bhh,
                    const float* __restrict__ cbih,
                    const float* __restrict__ cbhh,
                    const bf16_t* __restrict__ cWih_bf,
                    const bf16_t* __restrict__ cWhh_bf,
                    const bf16_t* __restrict__ Wih_bf,
                    const bf16_t* __restrict__ Whh_bf,
                    float* __restrict__ out) {
  __shared__ bf16_t h_new[ROWS * HSZ];                      // 16 KB (bf16 is lossless
                                                            //  here: only feeds A-frags)
  __shared__ __align__(16) bf16_t Bbuf[2][G3H * BPITCH];    // 2 x 15 KB double buffer
  __shared__ int typ_s[ROWS];
  __shared__ int counts[TNUM];
  __shared__ unsigned char rowlist[TNUM][ROWS];
  __shared__ int tiles[32];
  __shared__ int tileCount;

  const int tid  = threadIdx.x;
  const int lane = tid & 31;
  const int wid  = tid >> 5;
  const int l15  = lane & 15;
  const int hlf  = lane >> 4;
  const int aoff = hlf ? 8 : 0;    // A-frag K sub-offset (elements)
  const int boff = hlf ? 8 : 0;    // B-frag K offset (bf16)
  const long bb  = (long)blockIdx.x * ROWS;

  // ---------------- Phase 0: bucket rows by type ----------------
  if (tid < TNUM) counts[tid] = 0;
  if (tid == 0) tileCount = 0;
  if (tid < ROWS) typ_s[tid] = typ[bb + tid];
  __syncthreads();
  if (tid < ROWS) {
    int t = typ_s[tid];
    int idx = atomicAdd(&counts[t], 1);
    rowlist[t][idx] = (unsigned char)tid;
  }
  __syncthreads();
  if (tid == 0) {
    int tc = 0;
    for (int t = 0; t < TNUM; ++t)
      for (int m0 = 0; m0 < counts[t]; m0 += 16)
        tiles[tc++] = (t << 16) | m0;
    tileCount = tc;
  }
  __syncthreads();

  // ------- Phase 1: selected-type bank GRUCell via WMMA (one tile per wave) -------
  for (int ti = wid; ti < tileCount; ti += NWAVES) {
    const int t   = tiles[ti] >> 16;
    const int m0  = tiles[ti] & 0xffff;
    const int cnt = counts[t];
    const int mA  = m0 + l15;
    const int rA  = rowlist[t][(mA < cnt) ? mA : m0];   // clamp pad rows
    const long gA = bb + rA;
    const float*  xrow = x    + gA * ISZ;
    const float*  hrow = regs + gA * KCORE + t * HSZ;   // registers[b, t, :]
    const bf16_t* wih  = Wih_bf + (long)t * (G3H * ISZ);
    const bf16_t* whh  = Whh_bf + (long)t * (G3H * HSZ);

    v8f accS[8], accI[4], accN[4];  // r,z summed; in and hn kept apart
#pragma unroll
    for (int i = 0; i < 8; ++i) accS[i] = vzero8();
#pragma unroll
    for (int i = 0; i < 4; ++i) { accI[i] = vzero8(); accN[i] = vzero8(); }

#pragma unroll
    for (int kc = 0; kc < 2; ++kc) {
      v16bf ax = afrag_f32(xrow + kc * 32, aoff);
      v16bf ah = afrag_f32(hrow + kc * 32, aoff);
#pragma unroll
      for (int nt = 0; nt < 12; ++nt) {
        const int wofs = (nt * 16 + l15) * 64 + kc * 32 + boff;
        v16bf bi = bfrag_bf16(wih + wofs);
        v16bf bh = bfrag_bf16(whh + wofs);
        if (nt < 8) {                       // r,z gates: gi+gh sum directly
          accS[nt] = wmma_bf16(ax, bi, accS[nt]);
          accS[nt] = wmma_bf16(ah, bh, accS[nt]);
        } else {                            // n gate: keep in_/hn separate
          accI[nt - 8] = wmma_bf16(ax, bi, accI[nt - 8]);
          accN[nt - 8] = wmma_bf16(ah, bh, accN[nt - 8]);
        }
      }
    }

    const float* bih_t = bih + t * G3H;
    const float* bhh_t = bhh + t * G3H;
#pragma unroll
    for (int jt = 0; jt < 4; ++jt) {
      const int j = jt * 16 + l15;
      const float br  = bih_t[j] + bhh_t[j];
      const float bz  = bih_t[64 + j] + bhh_t[64 + j];
      const float bni = bih_t[128 + j];
      const float bnh = bhh_t[128 + j];
#pragma unroll
      for (int v = 0; v < 8; ++v) {
        const int mm   = m0 + v + hlf * 8;
        const bool ok  = (mm < cnt);
        const int  rm  = rowlist[t][ok ? mm : m0];
        const float hold = regs[(bb + rm) * KCORE + t * HSZ + j];
        const float r = sigm(accS[jt][v] + br);
        const float z = sigm(accS[4 + jt][v] + bz);
        const float n = tanhf(accI[jt][v] + bni + r * (accN[jt][v] + bnh));
        const float hv = (1.f - z) * n + z * hold;
        if (ok) h_new[rm * HSZ + j] = (bf16_t)hv;
      }
    }
  }
  __syncthreads();   // h_new complete for all 128 rows

  // ------- Phase 2: core GRUCell. cgi = reg_new @ cWih.T via WMMA, K=1024 -------
  {
    const int  r0    = wid * 16;
    const long gRow  = bb + r0 + l15;
    const int  myTyp = typ_s[r0 + l15];
    const float* aGlob = regs + gRow * KCORE;

    v8f accA[12];
#pragma unroll
    for (int i = 0; i < 12; ++i) accA[i] = vzero8();

    // prologue: stage K-slice 0 into buffer 0
    for (int g = tid; g < G3H; g += 256) {
      const bf16_t* src = cWih_bf + (long)g * KCORE;
      bf16_t* dst = &Bbuf[0][g * BPITCH];
#pragma unroll
      for (int q = 0; q < 4; ++q) cp16(src + q * 8, dst + q * 8);
    }
    wait_cp_barrier();

    for (int kc = 0; kc < 32; ++kc) {
      const bf16_t* cur = &Bbuf[kc & 1][0];
      // kick off the next K-slice into the other buffer (overlaps the 12 WMMAs)
      if (kc + 1 < 32) {
        bf16_t* nxt = &Bbuf[(kc + 1) & 1][0];
        for (int g = tid; g < G3H; g += 256) {
          const bf16_t* src = cWih_bf + (long)g * KCORE + (kc + 1) * 32;
#pragma unroll
          for (int q = 0; q < 4; ++q) cp16(src + q * 8, nxt + g * BPITCH + q * 8);
        }
        __builtin_prefetch(aGlob + (kc + 1) * 32, 0, 0);
      }

      // A fragment = registers row, with slot typ[row] substituted by h_new
      v16bf a;
      if (myTyp == (kc >> 1)) {
        a = afrag_bf16(h_new + (r0 + l15) * HSZ + (kc & 1) * 32, aoff);
      } else {
        a = afrag_f32(aGlob + kc * 32, aoff);
      }
#pragma unroll
      for (int nt = 0; nt < 12; ++nt) {
        v16bf b = bfrag_bf16(cur + (nt * 16 + l15) * BPITCH + boff);
        accA[nt] = wmma_bf16(a, b, accA[nt]);
      }
      wait_cp_barrier();
    }

    // cgh = c @ cWhh.T  (r,z summed into accA; hn separate)
    v8f accN2[4];
#pragma unroll
    for (int i = 0; i < 4; ++i) accN2[i] = vzero8();
    const float* crow = cvec + gRow * HSZ;
#pragma unroll
    for (int kc = 0; kc < 2; ++kc) {
      v16bf a = afrag_f32(crow + kc * 32, aoff);
#pragma unroll
      for (int nt = 0; nt < 12; ++nt) {
        v16bf b = bfrag_bf16(cWhh_bf + (nt * 16 + l15) * 64 + kc * 32 + boff);
        if (nt < 8) accA[nt] = wmma_bf16(a, b, accA[nt]);
        else        accN2[nt - 8] = wmma_bf16(a, b, accN2[nt - 8]);
      }
    }

    // fused GRU epilogue + coalesced store
#pragma unroll
    for (int jt = 0; jt < 4; ++jt) {
      const int j = jt * 16 + l15;
      const float br  = cbih[j] + cbhh[j];
      const float bz  = cbih[64 + j] + cbhh[64 + j];
      const float bni = cbih[128 + j];
      const float bnh = cbhh[128 + j];
#pragma unroll
      for (int v = 0; v < 8; ++v) {
        const long row = bb + r0 + v + hlf * 8;
        const float cold = cvec[row * HSZ + j];
        const float r = sigm(accA[jt][v] + br);
        const float z = sigm(accA[4 + jt][v] + bz);
        const float n = tanhf(accA[8 + jt][v] + bni + r * (accN2[jt][v] + bnh));
        out[row * HSZ + j] = (1.f - z) * n + z * cold;
      }
    }
  }
}

extern "C" void kernel_launch(void* const* d_in, const int* in_sizes, int n_in,
                              void* d_out, int out_size, void* d_ws, size_t ws_size,
                              hipStream_t stream) {
  const float* x    = (const float*)d_in[0];
  const int*   typ  = (const int*)  d_in[1];
  const float* c    = (const float*)d_in[2];
  const float* regs = (const float*)d_in[3];
  const float* Wih  = (const float*)d_in[4];
  const float* Whh  = (const float*)d_in[5];
  const float* bih  = (const float*)d_in[6];
  const float* bhh  = (const float*)d_in[7];
  const float* cWih = (const float*)d_in[8];
  const float* cWhh = (const float*)d_in[9];
  const float* cbih = (const float*)d_in[10];
  const float* cbhh = (const float*)d_in[11];
  (void)in_sizes; (void)n_in; (void)out_size; (void)ws_size;

  // bf16 weight mirrors in workspace (all 32B-aligned offsets)
  bf16_t* ws       = (bf16_t*)d_ws;
  bf16_t* cWih_bf  = ws;                         // 192*1024 = 196608
  bf16_t* cWhh_bf  = ws + 196608;                // 192*64   = 12288
  bf16_t* Wih_bf   = ws + 196608 + 12288;        // 16*192*64 = 196608
  bf16_t* Whh_bf   = Wih_bf + 196608;            // 16*192*64 = 196608

  cvt_bf16_kernel<<<(196608 + 255) / 256, 256, 0, stream>>>(cWih, cWih_bf, 196608);
  cvt_bf16_kernel<<<(12288  + 255) / 256, 256, 0, stream>>>(cWhh, cWhh_bf, 12288);
  cvt_bf16_kernel<<<(196608 + 255) / 256, 256, 0, stream>>>(Wih,  Wih_bf,  196608);
  cvt_bf16_kernel<<<(196608 + 255) / 256, 256, 0, stream>>>(Whh,  Whh_bf,  196608);

  mrn_gru_kernel<<<BATCH / ROWS, 256, 0, stream>>>(
      x, typ, c, regs, bih, bhh, cbih, cbhh,
      cWih_bf, cWhh_bf, Wih_bf, Whh_bf, (float*)d_out);
}
